// GCN_channel_68178310857300
// MI455X (gfx1250) — matrix operand order repaired and checked by reading, not compiled, not run on hardware
//
#include <hip/hip_runtime.h>

// MI455X / gfx1250, wave32. GEMM done via v_wmma_f32_16x16x32_f16 (f32 accum).
typedef __attribute__((ext_vector_type(16))) _Float16 v16h;
typedef __attribute__((ext_vector_type(8)))  float    v8f;

#define B_DIM   32
#define C_DIM   256
#define HW      4096
#define NTILE   128
// LDS row stride in halves: 256 data + 8 pad -> 528 bytes/row.
// 16B aligned (b128 fragment loads legal) and 132 dwords/row -> rows are 4
// banks apart, so the 16 rows touched by one half-wave cover all 64 banks.
#define KSTRIDE 264

#if defined(__has_builtin)
#if __has_builtin(__builtin_amdgcn_sched_group_barrier)
#define SCHED_GROUP(mask, size) __builtin_amdgcn_sched_group_barrier((mask), (size), 0)
#endif
#endif
#ifndef SCHED_GROUP
#define SCHED_GROUP(mask, size)
#endif

// ---------------- Kernel 1: c_adj[b][c] = mean(x[b,c,:,:]) ----------------
__global__ __launch_bounds__(256) void gcn_mean_kernel(
    const float* __restrict__ x, float* __restrict__ cadj) {
  __shared__ float red[256];
  const int row = blockIdx.x;          // b*C + c
  const int tid = threadIdx.x;
  const float4* xr = reinterpret_cast<const float4*>(x + (size_t)row * HW);
  float s = 0.f;
#pragma unroll
  for (int i = 0; i < 4; ++i) {        // 4 * 256 threads * float4 = 4096
    float4 v = xr[tid + i * 256];
    s += v.x + v.y + v.z + v.w;
  }
  red[tid] = s;
  __syncthreads();
  for (int off = 128; off > 0; off >>= 1) {
    if (tid < off) red[tid] += red[tid + off];
    __syncthreads();
  }
  if (tid == 0) cadj[row] = red[0] * (1.0f / HW);
}

// ---------------- Kernel 2: out[b] = relu((adj*s(c)) @ fea[b] * para) -----
// Block: (N-tile, batch). 256 threads = 8 waves.
__global__ __launch_bounds__(256) void gcn_gemm_kernel(
    const float* __restrict__ x, const float* __restrict__ adj,
    const float* __restrict__ para, const float* __restrict__ cadj,
    float* __restrict__ out) {
  extern __shared__ char smem[];
  _Float16* wlds = (_Float16*)smem;                  // [256][KSTRIDE] f16 ~132KB
  _Float16* flds = wlds + C_DIM * KSTRIDE;           // [NTILE][KSTRIDE] f16 ~66KB (transposed fea)
  float*    clds = (float*)(flds + NTILE * KSTRIDE); // [256] f32

  const int tid = threadIdx.x;
  const int b  = blockIdx.y;
  const int n0 = blockIdx.x * NTILE;

  // stage pooled means for this batch
  clds[tid] = cadj[b * C_DIM + tid];
  __syncthreads();

  // Build w[b] in f16: w[i][j] = adj[i][j] * 2*sigmoid(-|c_j - c_i|)
  // (the reference's abs/sigmoid expression is already symmetric)
  {
    const float ci = clds[tid];
    const float* adjrow = adj + (size_t)tid * C_DIM;
    unsigned int* wrow = (unsigned int*)(wlds + (size_t)tid * KSTRIDE);
#pragma unroll 4
    for (int j2 = 0; j2 < C_DIM / 2; ++j2) {
      const int j = 2 * j2;
      const float d0 = fabsf(clds[j]     - ci);
      const float d1 = fabsf(clds[j + 1] - ci);
      float s0 = (2.0f / (1.0f + __expf(d0))) * adjrow[j];
      float s1 = (2.0f / (1.0f + __expf(d1))) * adjrow[j + 1];
      union { unsigned int u; _Float16 h[2]; } p;
      p.h[0] = (_Float16)s0; p.h[1] = (_Float16)s1;
      wrow[j2] = p.u;
    }
  }

  // Stage fea tile fp32 -> f16, transposed: flds[n][k] = x[b, k, n0+n]
  {
    const int n     = tid & (NTILE - 1);
    const int khalf = tid >> 7;                      // 0 or 1
    const float* xb = x + (size_t)b * C_DIM * HW + n0 + n;
    unsigned int* frow = (unsigned int*)(flds + (size_t)n * KSTRIDE);
    const int k0 = khalf * 128;
#pragma unroll 4
    for (int k = k0; k < k0 + 128; k += 2) {         // coalesced across n
      const float a0 = xb[(size_t)k * HW];
      const float a1 = xb[(size_t)(k + 1) * HW];
      union { unsigned int u; _Float16 h[2]; } p;
      p.h[0] = (_Float16)a0; p.h[1] = (_Float16)a1;
      frow[k >> 1] = p.u;
    }
  }
  __syncthreads();

  // ---- WMMA main loop ----
  const int wave = tid >> 5;
  const int lane = tid & 31;
  const int lrow = lane & 15;
  const int hi   = lane >> 4;

  union Frag { v16h h; uint4 u[2]; };
  const v8f vzero = {};

  // B-fragment loader: 4 K-steps (8x ds_load_b128) into one buffer.
  // B layout: lane n<16 holds K 0-7 & 16-23 of column n; lanes 16-31 hold K 8-15 & 24-31.
  auto loadB4 = [&](Frag (&bf)[4], int nsub, int kq) {
    const _Float16* bbase = flds + (size_t)(nsub + lrow) * KSTRIDE + hi * 8;
#pragma unroll
    for (int k = 0; k < 4; ++k) {
      bf[k].u[0] = *reinterpret_cast<const uint4*>(bbase + (kq + k) * 32);
      bf[k].u[1] = *reinterpret_cast<const uint4*>(bbase + (kq + k) * 32 + 16);
    }
  };

#pragma unroll
  for (int mt = 0; mt < 2; ++mt) {
    const int mbase = wave * 32 + mt * 16;
    // A fragments (w rows), hoisted across the N loop: 8 K-steps x 8 VGPRs.
    // ISA 16-bit A layout: lanes0-15 hold K 0-7 & 16-23; lanes16-31 K 8-15 & 24-31.
    Frag afrag[8];
    {
      const _Float16* abase = wlds + (size_t)(mbase + lrow) * KSTRIDE + hi * 8;
#pragma unroll
      for (int k = 0; k < 8; ++k) {
        afrag[k].u[0] = *reinterpret_cast<const uint4*>(abase + k * 32);       // ds_load_b128
        afrag[k].u[1] = *reinterpret_cast<const uint4*>(abase + k * 32 + 16);  // ds_load_b128
      }
    }

    // Software-pipelined over 16 half-steps (half-step = 4 WMMAs of one
    // 16x16 output tile). Double-buffered B fragments: while 4 chained WMMAs
    // execute, the other buffer's 8 ds_load_b128 are in flight.
    Frag bbuf[2][4];
    loadB4(bbuf[0], 0, 0);
    v8f acc = vzero;
#pragma unroll
    for (int hs = 0; hs < 16; ++hs) {
      const int cur = hs & 1;
      if (hs < 15) {
        const int nh = hs + 1;
        loadB4(bbuf[nh & 1], (nh >> 1) * 16, (nh & 1) * 4);
      }
#pragma unroll
      for (int k = 0; k < 4; ++k) {
        acc = __builtin_amdgcn_wmma_f32_16x16x32_f16(
            false, afrag[cur * 4 + k].h, false, bbuf[cur][k].h,
            (short)0, acc, false, false);
      }
      // Pin schedule: 8 DS reads (next buffer) issue ahead of this half-step's
      // 4 WMMAs, per iteration.
      SCHED_GROUP(0x100, 8);  // DS read group
      SCHED_GROUP(0x008, 4);  // MFMA/WMMA group

      if (cur) {
        // Tile finished: epilogue out = relu(acc * para).
        // C/D layout: VGPR r -> row r (lanes0-15) or r+8 (lanes16-31); col = lane&15.
        const int nsub = (hs >> 1) * 16;
        const int ncol = n0 + nsub + lrow;
#pragma unroll
        for (int r = 0; r < 8; ++r) {
          const int mrow = mbase + r + hi * 8;
          const size_t pidx = (size_t)mrow * HW + ncol;
          const float o = acc[r] * para[pidx];
          out[(size_t)b * (C_DIM * (size_t)HW) + pidx] = o > 0.f ? o : 0.f;
        }
        acc = vzero;
      }
    }
  }
}

extern "C" void kernel_launch(void* const* d_in, const int* in_sizes, int n_in,
                              void* d_out, int out_size, void* d_ws, size_t ws_size,
                              hipStream_t stream) {
  const float* x    = (const float*)d_in[0];
  const float* adj  = (const float*)d_in[1];
  const float* para = (const float*)d_in[2];
  float* out  = (float*)d_out;
  float* cadj = (float*)d_ws;   // 8192 floats = 32 KB scratch

  gcn_mean_kernel<<<B_DIM * C_DIM, 256, 0, stream>>>(x, cadj);

  const size_t lds_bytes =
      (size_t)(C_DIM * KSTRIDE + NTILE * KSTRIDE) * sizeof(_Float16) +
      (size_t)C_DIM * sizeof(float);                // ~199 KB < 320 KB/WGP
  dim3 grid(HW / NTILE, B_DIM);
  gcn_gemm_kernel<<<grid, 256, lds_bytes, stream>>>(x, adj, para, cadj, out);
}
